// LocalitySelfAttention_18511309046452
// MI455X (gfx1250) — matrix operand order repaired and checked by reading, not compiled, not run on hardware
//
#include <hip/hip_runtime.h>

// ---------------------------------------------------------------------------
// LocalitySelfAttention for MI455X (gfx1250, wave32, WMMA f16 16x16x32)
// B=2, N=4096, C=256, H=8, d=32. Flash-attention: [B,H,N,N] never hits memory.
// f16 operands everywhere (pure vector loads feeding v_wmma_f32_16x16x32_f16);
// online-softmax stats batched over 64-key tiles to cut VALU bookkeeping.
// ---------------------------------------------------------------------------

typedef __attribute__((ext_vector_type(16))) _Float16 v16h;
typedef __attribute__((ext_vector_type(8)))  _Float16 v8h;
typedef __attribute__((ext_vector_type(4)))  _Float16 v4h;
typedef __attribute__((ext_vector_type(8)))  float    v8f;

#define NUM_HEADS 8
#define DIM       256
#define HEAD_DIM  32
#define SEQ       4096
#define BATCH     2
#define ROWS      (BATCH * SEQ)        // 8192

// --- WMMA f16 16x16x32 fragment index maps used here (contiguous k, wave32) ------
// A (16x32, MxK): lane m = L&15; halves at k = hi*16 + {0..15}
//                 (hi = L>=16) -> two contiguous 16B chunks per lane.
// B (32x16, KxN): lane n = L&15; halves at k = hi*16 + {0..15}
//                 -> one contiguous 32B chunk per lane.
// C/D (16x16 f32): VGPR r -> row M = r + hi*8, column N = L&15.

static __device__ __forceinline__ v8f wmma_f16(v16h a, v16h b, v8f c) {
  return __builtin_amdgcn_wmma_f32_16x16x32_f16(false, a, false, b,
                                                (short)0, c, false, false);
}

static __device__ __forceinline__ v16h load_a_frag(const _Float16* row,
                                                   int k0, int hi) {
  const v8h lo = *(const v8h*)(row + k0 + hi * 16);
  const v8h hh = *(const v8h*)(row + k0 + hi * 16 + 8);
  v16h a;
#pragma unroll
  for (int e = 0; e < 8; ++e) { a[e] = lo[e]; a[8 + e] = hh[e]; }
  return a;
}

// ---------------------------------------------------------------------------
// Kernel 0: f32 -> f16 elementwise conversion, 4 elems/thread.
// ---------------------------------------------------------------------------
__global__ __launch_bounds__(256) void cvt_f16_kernel(
    const float* __restrict__ s, _Float16* __restrict__ d, int n4) {
  const int i = blockIdx.x * 256 + threadIdx.x;
  if (i >= n4) return;
  const float4 v = ((const float4*)s)[i];
  v4h h;
  h[0] = (_Float16)v.x; h[1] = (_Float16)v.y;
  h[2] = (_Float16)v.z; h[3] = (_Float16)v.w;
  ((v4h*)d)[i] = h;
}

// ---------------------------------------------------------------------------
// Kernel 1: qkv = x @ w_qkv^T (f16 in, f32 acc), scattered into
//   q [bh][n][d], k [bh][n][d], v TRANSPOSED [bh][d][n] (for P@V B-frags).
// One wave = 16x64 output chunk. 512 Mtiles * 12 Nchunks = 6144 waves.
// ---------------------------------------------------------------------------
__global__ __launch_bounds__(128) void qkv_gemm_kernel(
    const _Float16* __restrict__ xf, const _Float16* __restrict__ wf,
    _Float16* __restrict__ qf, _Float16* __restrict__ kf,
    _Float16* __restrict__ vt) {
  const int wave    = blockIdx.x * 4 + (threadIdx.x >> 5);
  const int mtile   = wave / 12;
  const int colbase = (wave % 12) * 64;
  const int mbase   = mtile * 16;
  const int lane    = threadIdx.x & 31;
  const int hi      = lane >> 4;
  const int lm      = lane & 15;

  const _Float16* arow = xf + (size_t)(mbase + lm) * DIM;
  v8f acc[4] = {};
  for (int ks = 0; ks < 8; ++ks) {          // K = 256 = 8 * 32
    const int k0 = ks * 32;
    const v16h a = load_a_frag(arow, k0, hi);
#pragma unroll
    for (int t = 0; t < 4; ++t) {
      const v16h b = *(const v16h*)(wf + (size_t)(colbase + 16 * t + lm) * DIM
                                    + k0 + hi * 16);
      acc[t] = wmma_f16(a, b, acc[t]);
    }
  }
#pragma unroll
  for (int t = 0; t < 4; ++t) {
#pragma unroll
    for (int r = 0; r < 8; ++r) {
      const int row = mbase + r + hi * 8;   // token row (0..8191)
      const int c   = colbase + 16 * t + lm;
      const int mat = c >> 8;               // 0=q 1=k 2=v
      const int rem = c & 255;
      const int h   = rem >> 5;
      const int dc  = rem & 31;
      const int b_  = row >> 12;
      const int nn  = row & 4095;
      const int bh  = b_ * NUM_HEADS + h;
      const _Float16 val = (_Float16)acc[t][r];
      if (mat == 0)
        qf[((size_t)bh * SEQ + nn) * HEAD_DIM + dc] = val;
      else if (mat == 1)
        kf[((size_t)bh * SEQ + nn) * HEAD_DIM + dc] = val;
      else
        vt[((size_t)bh * HEAD_DIM + dc) * SEQ + nn] = val;  // transposed
    }
  }
}

// ---------------------------------------------------------------------------
// Kernel 2: flash attention. One wave = 16 query rows of one (b,h).
// Keys in tiles of 64: 4 score WMMAs -> ONE batched online-softmax update
// -> LDS C->A transpose of P(16x64) -> 4 P@V WMMAs (K=64 in 2 steps).
// 16 bh * 256 qtiles = 4096 waves.
// ---------------------------------------------------------------------------
__global__ __launch_bounds__(128) void attn_kernel(
    const _Float16* __restrict__ qf, const _Float16* __restrict__ kf,
    const _Float16* __restrict__ vt, const float* __restrict__ temperature,
    _Float16* __restrict__ of) {
  __shared__ _Float16 pbuf[4][16][64];      // 2 KB per wave P staging
  const int wid   = threadIdx.x >> 5;
  const int wave  = blockIdx.x * 4 + wid;
  const int bh    = wave >> 8;              // 0..15
  const int qtile = wave & 255;             // 0..255
  const int lane  = threadIdx.x & 31;
  const int hi    = lane >> 4;
  const int lm    = lane & 15;
  const float invt  = 1.0f / temperature[0];
  const float scale = 0.0625f;              // 256^-0.5

  const _Float16* qb  = qf + (size_t)bh * SEQ * HEAD_DIM;
  const _Float16* kb  = kf + (size_t)bh * SEQ * HEAD_DIM;
  const _Float16* vtb = vt + (size_t)bh * HEAD_DIM * SEQ;
  const int q0 = qtile * 16;

  const v16h aq = load_a_frag(qb + (size_t)(q0 + lm) * HEAD_DIM, 0, hi);

  v8f acc0 = {}, acc1 = {};                 // O accumulators (16x32 f32)
  float m_i[8], l_i[8];
#pragma unroll
  for (int r = 0; r < 8; ++r) { m_i[r] = -3.0e38f; l_i[r] = 0.0f; }

  for (int j = 0; j < SEQ; j += 64) {
    // prefetch next key tile (K rows + V^T cols)
    if (j + 64 < SEQ) {
      __builtin_prefetch(kb + (size_t)(j + 64 + lm) * HEAD_DIM, 0, 0);
      __builtin_prefetch(vtb + (size_t)lm * SEQ + j + 64, 0, 0);
    }
    // ---- scores: S = Q @ K^T for keys [j, j+64) ----
    const v16h bk0 =
        *(const v16h*)(kb + (size_t)(j + lm) * HEAD_DIM + hi * 16);
    const v16h bk1 =
        *(const v16h*)(kb + (size_t)(j + 16 + lm) * HEAD_DIM + hi * 16);
    const v16h bk2 =
        *(const v16h*)(kb + (size_t)(j + 32 + lm) * HEAD_DIM + hi * 16);
    const v16h bk3 =
        *(const v16h*)(kb + (size_t)(j + 48 + lm) * HEAD_DIM + hi * 16);
    const v8f z = {};
    v8f s0 = wmma_f16(aq, bk0, z);
    v8f s1 = wmma_f16(aq, bk1, z);
    v8f s2 = wmma_f16(aq, bk2, z);
    v8f s3 = wmma_f16(aq, bk3, z);

    // ---- scale, diagonal +1e9 boost, /temperature (in place) ----
#pragma unroll
    for (int r = 0; r < 8; ++r) {
      const int qrow = q0 + r + hi * 8;
      s0[r] = (s0[r] * scale + ((qrow == (j + lm))      ? 1.0e9f : 0.0f)) * invt;
      s1[r] = (s1[r] * scale + ((qrow == (j + 16 + lm)) ? 1.0e9f : 0.0f)) * invt;
      s2[r] = (s2[r] * scale + ((qrow == (j + 32 + lm)) ? 1.0e9f : 0.0f)) * invt;
      s3[r] = (s3[r] * scale + ((qrow == (j + 48 + lm)) ? 1.0e9f : 0.0f)) * invt;
    }
    // ---- batched online softmax: one stats update per 64 keys ----
#pragma unroll
    for (int r = 0; r < 8; ++r) {
      float mx = fmaxf(fmaxf(s0[r], s1[r]), fmaxf(s2[r], s3[r]));
      mx = fmaxf(mx, __shfl_xor(mx, 1, 32));
      mx = fmaxf(mx, __shfl_xor(mx, 2, 32));
      mx = fmaxf(mx, __shfl_xor(mx, 4, 32));
      mx = fmaxf(mx, __shfl_xor(mx, 8, 32));
      const float mnew  = fmaxf(m_i[r], mx);
      const float alpha = __expf(m_i[r] - mnew);
      m_i[r] = mnew;
      s0[r] = __expf(s0[r] - mnew);
      s1[r] = __expf(s1[r] - mnew);
      s2[r] = __expf(s2[r] - mnew);
      s3[r] = __expf(s3[r] - mnew);
      float rs = (s0[r] + s1[r]) + (s2[r] + s3[r]);
      rs += __shfl_xor(rs, 1, 32);
      rs += __shfl_xor(rs, 2, 32);
      rs += __shfl_xor(rs, 4, 32);
      rs += __shfl_xor(rs, 8, 32);
      l_i[r] = l_i[r] * alpha + rs;
      acc0[r] *= alpha;
      acc1[r] *= alpha;
    }
    // ---- P(16x64): C-layout -> A-layout via per-wave LDS staging ----
#pragma unroll
    for (int r = 0; r < 8; ++r) {
      const int m = r + hi * 8;
      pbuf[wid][m][lm]      = (_Float16)s0[r];
      pbuf[wid][m][16 + lm] = (_Float16)s1[r];
      pbuf[wid][m][32 + lm] = (_Float16)s2[r];
      pbuf[wid][m][48 + lm] = (_Float16)s3[r];
    }
    asm volatile("s_wait_dscnt 0" ::: "memory");   // wave-local LDS RAW
    const v16h ap0 = load_a_frag(&pbuf[wid][lm][0], 0, hi);
    const v16h ap1 = load_a_frag(&pbuf[wid][lm][0], 32, hi);

    // ---- O += P @ V : B(k,n) = vt[n][j+k], contiguous in k ----
    const v16h bv00 = *(const v16h*)(vtb + (size_t)lm * SEQ + j + hi * 16);
    const v16h bv01 = *(const v16h*)(vtb + (size_t)lm * SEQ + j + 32 + hi * 16);
    const v16h bv10 =
        *(const v16h*)(vtb + (size_t)(16 + lm) * SEQ + j + hi * 16);
    const v16h bv11 =
        *(const v16h*)(vtb + (size_t)(16 + lm) * SEQ + j + 32 + hi * 16);
    acc0 = wmma_f16(ap0, bv00, acc0);
    acc0 = wmma_f16(ap1, bv01, acc0);
    acc1 = wmma_f16(ap0, bv10, acc1);
    acc1 = wmma_f16(ap1, bv11, acc1);
  }

  // ---- normalize, store O as f16 in [B][N][H*d] (head-concat) layout ----
  const int b_ = bh >> 3;
  const int h  = bh & 7;
#pragma unroll
  for (int r = 0; r < 8; ++r) {
    const float inv = 1.0f / l_i[r];
    const int nq = q0 + r + hi * 8;
    const size_t rowbase = ((size_t)(b_ * SEQ + nq)) * DIM + h * HEAD_DIM;
    of[rowbase + lm]      = (_Float16)(acc0[r] * inv);
    of[rowbase + 16 + lm] = (_Float16)(acc1[r] * inv);
  }
}

// ---------------------------------------------------------------------------
// Kernel 3: out = o @ w_proj^T + b_proj, f32 output.
// 512 Mtiles * 4 Nchunks = 2048 waves.
// ---------------------------------------------------------------------------
__global__ __launch_bounds__(128) void proj_kernel(
    const _Float16* __restrict__ of, const _Float16* __restrict__ wf,
    const float* __restrict__ b_proj, float* __restrict__ out) {
  const int wave    = blockIdx.x * 4 + (threadIdx.x >> 5);
  const int mtile   = wave >> 2;
  const int colbase = (wave & 3) * 64;
  const int mbase   = mtile * 16;
  const int lane    = threadIdx.x & 31;
  const int hi      = lane >> 4;
  const int lm      = lane & 15;

  const _Float16* arow = of + (size_t)(mbase + lm) * DIM;
  v8f acc[4] = {};
  for (int ks = 0; ks < 8; ++ks) {
    const int k0 = ks * 32;
    const v16h a = load_a_frag(arow, k0, hi);
#pragma unroll
    for (int t = 0; t < 4; ++t) {
      const v16h b = *(const v16h*)(wf + (size_t)(colbase + 16 * t + lm) * DIM
                                    + k0 + hi * 16);
      acc[t] = wmma_f16(a, b, acc[t]);
    }
  }
#pragma unroll
  for (int t = 0; t < 4; ++t) {
    const int c = colbase + 16 * t + lm;
    const float bias = b_proj[c];
#pragma unroll
    for (int r = 0; r < 8; ++r) {
      const int row = mbase + r + hi * 8;
      out[(size_t)row * DIM + c] = acc[t][r] + bias;
    }
  }
}

// ---------------------------------------------------------------------------
extern "C" void kernel_launch(void* const* d_in, const int* in_sizes, int n_in,
                              void* d_out, int out_size, void* d_ws,
                              size_t ws_size, hipStream_t stream) {
  const float* x      = (const float*)d_in[0];
  const float* w_qkv  = (const float*)d_in[1];
  const float* w_proj = (const float*)d_in[2];
  const float* b_proj = (const float*)d_in[3];
  const float* temp   = (const float*)d_in[4];
  float* out = (float*)d_out;

  // Workspace layout (f16 buffers):
  //  [ 0,  4M) xf      8192x256
  //  [ 4M, 8M) qf      [16][4096][32]
  //  [ 8M,12M) kf      [16][4096][32]
  //  [12M,16M) vt      [16][32][4096]   (V transposed)
  //  [16M,20M) of      8192x256
  //  [20M, +384K) wqkvf 768x256
  //  [20M+512K, +128K) wprojf 256x256
  char* ws = (char*)d_ws;
  _Float16* xf     = (_Float16*)(ws);
  _Float16* qf     = (_Float16*)(ws + ((size_t)4  << 20));
  _Float16* kf     = (_Float16*)(ws + ((size_t)8  << 20));
  _Float16* vt     = (_Float16*)(ws + ((size_t)12 << 20));
  _Float16* of     = (_Float16*)(ws + ((size_t)16 << 20));
  _Float16* wqkvf  = (_Float16*)(ws + ((size_t)20 << 20));
  _Float16* wprojf = (_Float16*)(ws + ((size_t)20 << 20) + ((size_t)512 << 10));

  cvt_f16_kernel<<<2048, 256, 0, stream>>>(x, xf, ROWS * DIM / 4);
  cvt_f16_kernel<<<192,  256, 0, stream>>>(w_qkv, wqkvf, 3 * DIM * DIM / 4);
  cvt_f16_kernel<<<64,   256, 0, stream>>>(w_proj, wprojf, DIM * DIM / 4);

  // 512 Mtiles * 12 Nchunks = 6144 waves / 4 per block
  qkv_gemm_kernel<<<1536, 128, 0, stream>>>(xf, wqkvf, qf, kf, vt);
  // 16 (b,h) * 256 qtiles = 4096 waves / 4 per block
  attn_kernel<<<1024, 128, 0, stream>>>(qf, kf, vt, temp, of);
  // 512 Mtiles * 4 Nchunks = 2048 waves / 4 per block
  proj_kernel<<<512, 128, 0, stream>>>(of, wprojf, b_proj, out);
}
